// imperf_coil_lowrank_41832981463230
// MI455X (gfx1250) — matrix operand order repaired and checked by reading, not compiled, not run on hardware
//
#include <hip/hip_runtime.h>
#include <hip/hip_fp16.h>

typedef _Float16 v16h __attribute__((ext_vector_type(16)));
typedef float    v8f  __attribute__((ext_vector_type(8)));
typedef _Float16 h2   __attribute__((ext_vector_type(2)));
typedef unsigned int v4u __attribute__((ext_vector_type(4)));
typedef int      v8i  __attribute__((ext_vector_type(8)));
typedef int      v4i  __attribute__((ext_vector_type(4)));

#define H_ 128
#define W_ 128
#define K_ 32768
#define L_ 6
#define C_ 12

static constexpr float TWO_PI_F = 6.28318530717958647692f;

// LDS carve (dynamic shared):
//   [0, 128K)        : B double buffer, 2 x 32768 halves (64 KB each)
//   [128K, 192K)     : Ey tiles, 8 waves x 16 x 128 h2        (64 KB)
//   [192K, +6K)      : FBx, 8 waves x 6 x 16 float2            (6 KB)
#define LDS_B_OFF   0
#define LDS_EY_OFF  131072
#define LDS_FBX_OFF 196608
#define LDS_TOTAL   202752

// ---------------------------------------------------------------------------
// Prep kernel: Bx[l,x,y] = (sf + i sf) * (img + i img) -> f16, stored in WMMA
// B-fragment order, grouped per segment so one contiguous 64 KB block per l:
//   pB layout: [l][plane(re=0,im=1)][nt(8)][kc(4)][lane(32)][h(16)] halves.
// B (32x16) layout: lanes 0-15 h<8->K=h, h>=8->K=16+h-8; lanes 16-31 h<8->K=8+h, h>=8->K=24+h-8; N=lane&15.
// ---------------------------------------------------------------------------
__global__ void pack_b_kernel(const float* __restrict__ img_r,
                              const float* __restrict__ img_i,
                              const float* __restrict__ sf_r,
                              const float* __restrict__ sf_i,
                              _Float16* __restrict__ pB) {
  int idx = blockIdx.x * blockDim.x + threadIdx.x;  // 6*2*16384 = 196608
  if (idx >= L_ * 2 * 16384) return;
  int h     = idx & 15;
  int lane  = (idx >> 4) & 31;
  int kc    = (idx >> 9) & 3;
  int nt    = (idx >> 11) & 7;
  int plane = (idx >> 14) & 1;
  int l     = idx >> 15;
  int Kk    = ((lane & 16) ? 8 : 0) + (h & 7) + ((h & 8) ? 16 : 0);
  int x     = kc * 32 + Kk;
  int y     = nt * 16 + (lane & 15);
  float ir = img_r[x * W_ + y], ii = img_i[x * W_ + y];
  float sr = sf_r[(l * H_ + x) * W_ + y], si = sf_i[(l * H_ + x) * W_ + y];
  float v  = plane ? (sr * ii + si * ir) : (sr * ir - si * ii);
  pB[idx] = (_Float16)v;
}

// ---------------------------------------------------------------------------
// TDM issue: 1-D copy of one 64 KB segment block (8192 x 8B elements) from
// global pB into LDS at byte offset ldsOff.  D# built per CDNA5 ISA §8.
// ---------------------------------------------------------------------------
__device__ __forceinline__ void tdm_load_segment(const _Float16* gsrc,
                                                 unsigned ldsOff) {
  unsigned long long ga = (unsigned long long)(uintptr_t)gsrc;
  v4u g0;
  g0[0] = 1u;                                   // count=1, user mode
  g0[1] = ldsOff;                               // lds_addr (bytes)
  g0[2] = (unsigned)(ga & 0xffffffffu);         // global_addr[31:0]
  g0[3] = (unsigned)((ga >> 32) & 0x01ffffffu)  // global_addr[56:32]
          | 0x80000000u;                        // type=2 ("image")
  v8i g1;
  g1[0] = 0x00030000;      // wg_mask=0, data_size=3 (8B)
  g1[1] = 0x20000000;      // tensor_dim0[15:0]=8192 in bits[31:16]
  g1[2] = 0x00010000;      // tensor_dim0[31:16]=0, tensor_dim1=1
  g1[3] = 0x20000000;      // tile_dim0=8192 in bits[31:16]
  g1[4] = 0;               // tile_dim1=0 (unused), tile_dim2=0
  g1[5] = 8192;            // tensor_dim0_stride[31:0]
  g1[6] = 0;
  g1[7] = 0;
  v4i z4 = {};
  v8i z8 = {};
  __builtin_amdgcn_tensor_load_to_lds(g0, g1, z4, z4, z8, 0);
}

// ---------------------------------------------------------------------------
// Main fused NUDFT kernel. One wave = 16 k-samples; 8 waves per block.
// ---------------------------------------------------------------------------
__global__ __launch_bounds__(256, 1) void nudft_wmma_kernel(
    const float* __restrict__ trj,      // (K,2)
    const _Float16* __restrict__ pB,    // packed B, [l][plane][frag]
    const float* __restrict__ tf_r,     // (L,C,K)
    const float* __restrict__ tf_i,     // (L,C,K)
    float* __restrict__ out) {          // (C,K,2)
  extern __shared__ char smem[];
  _Float16* bLds = (_Float16*)(smem + LDS_B_OFF);
  h2*       eyT  = (h2*)(smem + LDS_EY_OFF);
  float2*   fbxT = (float2*)(smem + LDS_FBX_OFF);
  const unsigned ldsBase = (unsigned)(uintptr_t)smem;

  const int lane = threadIdx.x & 31;
  const int wv   = threadIdx.x >> 5;
  const int k0   = blockIdx.x * 128 + wv * 16;

  // kick off TDM pipeline (wave 0 only; TDM ignores EXEC, descriptor uniform)
  if (wv == 0) {
    tdm_load_segment(pB + 0 * 32768, ldsBase + LDS_B_OFF + 0 * 65536);
    tdm_load_segment(pB + 1 * 32768, ldsBase + LDS_B_OFF + 1 * 65536);
  }

  // ---- Phase 1: Ey[k, y] = exp(-2*pi*i * trj[k,1] * (y-64)/128) into LDS --
  #pragma unroll 4
  for (int j = 0; j < 64; ++j) {
    int flat = lane * 64 + j;
    int m = flat >> 7;
    int y = flat & 127;
    float ty = trj[(k0 + m) * 2 + 1];
    float ph = -TWO_PI_F * ty * ((float)(y - 64) * (1.0f / 128.0f));
    float s, c;
    __sincosf(ph, &s, &c);
    h2 e;
    e.x = (_Float16)c;
    e.y = (_Float16)s;
    eyT[(wv * 16 + m) * 128 + y] = e;
  }

  // ---- Phase 2: Ex A-fragments straight into WMMA register layout --------
  const int mrow  = lane & 15;
  const int half  = (lane >> 4);
  const int kAoff = half ? 8 : 0;
  const float tx  = trj[(k0 + mrow) * 2 + 0];

  v16h a_r[4], a_i[4];
  #pragma unroll
  for (int kc = 0; kc < 4; ++kc) {
    #pragma unroll
    for (int h = 0; h < 16; ++h) {
      int Kk = kAoff + (h & 7) + ((h & 8) ? 16 : 0);
      int x  = kc * 32 + Kk;
      float ph = -TWO_PI_F * tx * ((float)(x - 64) * (1.0f / 128.0f));
      float s, c;
      __sincosf(ph, &s, &c);
      a_r[kc][h] = (_Float16)c;
      a_i[kc][h] = (_Float16)s;
    }
  }

  // ---- Phase 3: per segment l: complex GEMM (B from LDS) + Ey contraction -
  #pragma unroll
  for (int l = 0; l < L_; ++l) {
    if (wv == 0) {
      if (l < 5) {
        __builtin_amdgcn_s_wait_tensorcnt(1);   // buf[l&1] load complete
      } else {
        __builtin_amdgcn_s_wait_tensorcnt(0);   // last buffer
      }
    }
    __syncthreads();

    float fbr[8] = {0.f, 0.f, 0.f, 0.f, 0.f, 0.f, 0.f, 0.f};
    float fbi[8] = {0.f, 0.f, 0.f, 0.f, 0.f, 0.f, 0.f, 0.f};
    const _Float16* bR0 = bLds + (l & 1) * 32768;            // re plane
    const _Float16* bI0 = bR0 + 16384;                       // im plane

    for (int nt = 0; nt < 8; ++nt) {
      v8f acc_rr = {};
      v8f acc_ii = {};
      v8f acc_ri = {};
      #pragma unroll
      for (int kc = 0; kc < 4; ++kc) {
        int foff = ((nt * 4 + kc) * 32 + lane) * 16;
        v16h br = *(const v16h*)(bR0 + foff);
        v16h bi = *(const v16h*)(bI0 + foff);
        acc_rr = __builtin_amdgcn_wmma_f32_16x16x32_f16(
            false, a_r[kc], false, br, (short)0, acc_rr, false, false);
        acc_ii = __builtin_amdgcn_wmma_f32_16x16x32_f16(
            false, a_i[kc], false, bi, (short)0, acc_ii, false, false);
        acc_ri = __builtin_amdgcn_wmma_f32_16x16x32_f16(
            false, a_r[kc], false, bi, (short)0, acc_ri, false, false);
        acc_ri = __builtin_amdgcn_wmma_f32_16x16x32_f16(
            false, a_i[kc], false, br, (short)0, acc_ri, false, false);
      }
      int y = nt * 16 + (lane & 15);
      #pragma unroll
      for (int r = 0; r < 8; ++r) {
        float cr = acc_rr[r] - acc_ii[r];
        float ci = acc_ri[r];
        int m = r + (half ? 8 : 0);
        h2 e = eyT[(wv * 16 + m) * 128 + y];
        float er = (float)e.x, ei = (float)e.y;
        fbr[r] += cr * er - ci * ei;
        fbi[r] += cr * ei + ci * er;
      }
    }

    #pragma unroll
    for (int r = 0; r < 8; ++r) {
      #pragma unroll
      for (int mask = 1; mask < 16; mask <<= 1) {
        fbr[r] += __shfl_xor(fbr[r], mask, 16);
        fbi[r] += __shfl_xor(fbi[r], mask, 16);
      }
    }
    if ((lane & 15) == 0) {
      int kh = half * 8;
      #pragma unroll
      for (int r = 0; r < 8; ++r)
        fbxT[(wv * L_ + l) * 16 + kh + r] = make_float2(fbr[r], fbi[r]);
    }

    __syncthreads();                       // everyone done reading buf[l&1]
    if (wv == 0 && l < 4)                  // refill it for segment l+2
      tdm_load_segment(pB + (l + 2) * 32768,
                       ldsBase + LDS_B_OFF + (l & 1) * 65536);
  }

  // ---- Phase 4: ksp[c,k] = sum_l FBx[l,k] * tf[l,c,k]  (same-wave LDS) ---
  const int kk = lane & 15;
  const int kg = k0 + kk;
  const int c0 = half * 6;
  #pragma unroll
  for (int cl = 0; cl < 6; ++cl) {
    int c = c0 + cl;
    float ar = 0.f, ai = 0.f;
    #pragma unroll
    for (int l = 0; l < L_; ++l) {
      float2 fb = fbxT[(wv * L_ + l) * 16 + kk];
      float tr = tf_r[((size_t)(l * C_ + c)) * K_ + kg];
      float ti = tf_i[((size_t)(l * C_ + c)) * K_ + kg];
      ar += fb.x * tr - fb.y * ti;
      ai += fb.x * ti + fb.y * tr;
    }
    out[((size_t)c * K_ + kg) * 2 + 0] = ar;
    out[((size_t)c * K_ + kg) * 2 + 1] = ai;
  }
}

extern "C" void kernel_launch(void* const* d_in, const int* in_sizes, int n_in,
                              void* d_out, int out_size, void* d_ws, size_t ws_size,
                              hipStream_t stream) {
  const float* img_r = (const float*)d_in[0];
  const float* img_i = (const float*)d_in[1];
  const float* sf_r  = (const float*)d_in[2];
  const float* sf_i  = (const float*)d_in[3];
  const float* tf_r  = (const float*)d_in[4];
  const float* tf_i  = (const float*)d_in[5];
  const float* trj   = (const float*)d_in[6];

  const int packElems = L_ * 2 * 16384;       // 196608 halves = 384 KB in d_ws
  _Float16* pB = (_Float16*)d_ws;

  (void)hipFuncSetAttribute((const void*)nudft_wmma_kernel,
                            hipFuncAttributeMaxDynamicSharedMemorySize,
                            LDS_TOTAL);

  pack_b_kernel<<<(packElems + 255) / 256, 256, 0, stream>>>(
      img_r, img_i, sf_r, sf_i, pB);

  nudft_wmma_kernel<<<K_ / 128, 256, LDS_TOTAL, stream>>>(
      trj, pB, tf_r, tf_i, (float*)d_out);
}
